// Evoformer_57586921505289
// MI455X (gfx1250) — matrix operand order repaired and checked by reading, not compiled, not run on hardware
//
#include <hip/hip_runtime.h>
#include <hip/hip_bf16.h>

// ---------------------------------------------------------------------------
// Evoformer MSA row attention w/ pair bias — gfx1250 (CDNA5, wave32, WMMA)
// All GEMMs run on v_wmma_f32_16x16x32_bf16. Intermediates kept bf16.
// Async LDS staging (GLOBAL_LOAD_ASYNC_TO_LDS_B128) used where copies are
// conversion-free, guarded by __has_builtin for toolchain portability.
// ---------------------------------------------------------------------------

typedef __bf16 bf16_t;
typedef __attribute__((ext_vector_type(16))) __bf16 v16bf;
typedef __attribute__((ext_vector_type(8)))  float  v8f;
typedef __attribute__((ext_vector_type(4)))  int    v4i;

#define DEV __device__ __forceinline__

DEV v8f wmma_bf16(v16bf a, v16bf b, v8f c) {
  // (neg_a, A, neg_b, B, c_mod, C, reuse_a, reuse_b)
  return __builtin_amdgcn_wmma_f32_16x16x32_bf16(false, a, false, b, (short)0, c,
                                                 false, false);
}

// ---------------- async (ASYNCcnt) LDS staging, 16B per lane ---------------
#if __has_builtin(__builtin_amdgcn_global_load_async_to_lds_b128)
#define HAVE_ASYNC_COPY 1
typedef __attribute__((address_space(1))) v4i* gv4i_p;
typedef __attribute__((address_space(3))) v4i* lv4i_p;
DEV void async_copy16(void* dst_lds, const void* src_glob) {
  __builtin_amdgcn_global_load_async_to_lds_b128(
      (gv4i_p)src_glob, (lv4i_p)dst_lds, 0, 0);
}
DEV void async_wait0() {
#if __has_builtin(__builtin_amdgcn_s_wait_asynccnt)
  __builtin_amdgcn_s_wait_asynccnt(0);
#else
  asm volatile("s_wait_asynccnt 0" ::: "memory");
#endif
}
#else
#define HAVE_ASYNC_COPY 0
DEV void async_wait0() {}
#endif

// A fragment: 16x32 (MxK) bf16, rows of `base` with stride lda.
// Lanes 0-15 hold M=lane, K 0..15; lanes 16-31 K 16..31 (matches B fragments).
DEV v16bf load_a_frag(const bf16_t* base, int lda, int lane) {
  const int m  = lane & 15;
  const int kb = (lane >> 4) * 16;
  const bf16_t* p = base + (size_t)m * lda;
  v16bf a;
#pragma unroll
  for (int e = 0; e < 8; ++e)  a[e] = p[kb + e];        // K = kb + e
#pragma unroll
  for (int e = 8; e < 16; ++e) a[e] = p[kb + e];        // K = kb + e
  return a;
}

// B fragment: 32x16 (KxN) bf16 from a K-major array (stride ldb between K rows).
// Lanes 0-15 hold N=lane, K=0..15; lanes 16-31 hold K=16..31.
DEV v16bf load_b_frag(const bf16_t* base, int ldb, int lane) {
  const int n  = lane & 15;
  const int kb = (lane >> 4) * 16;
  v16bf b;
#pragma unroll
  for (int e = 0; e < 16; ++e) b[e] = base[(size_t)(kb + e) * ldb + n];
  return b;
}

// B fragment where logical B = Xt^T, Xt stored n-major ([n][k], row stride ld).
// Element (k, n) = base[n*ld + k] -> contiguous 16 bf16 per lane (2x ds_load_b128).
DEV v16bf load_bT_frag(const bf16_t* base, int ld, int lane) {
  const int n  = lane & 15;
  const int kb = (lane >> 4) * 16;
  const bf16_t* p = base + (size_t)n * ld + kb;
  v16bf b;
#pragma unroll
  for (int e = 0; e < 16; ++e) b[e] = p[e];
  return b;
}

// ---------------------------------------------------------------------------
// Generic GEMM: Y[M,N] = op(A[M,K] * B[K,N]).  A: f32 or bf16, B: f32.
// Block: 256 threads (8 waves). Tile: 64(M) x 64(N), K-chunk 32.
// Each wave owns two N-adjacent 16x16 tiles sharing one A fragment
// (2 WMMAs / 3 fragment loads per chunk).
// ---------------------------------------------------------------------------
template <typename AT, typename OT, bool SIGMOID>
__global__ __launch_bounds__(256) void gemm_wmma_kernel(
    const AT* __restrict__ A, const float* __restrict__ B, OT* __restrict__ Y,
    int M, int N, int K) {
  constexpr int BM = 64, BN = 64, BK = 32;
  constexpr bool A_BF16 = (sizeof(AT) == 2);
  __shared__ __align__(16) bf16_t As[2][BM][BK];
  __shared__ __align__(16) bf16_t Bs[2][BK][BN];

  const int tid  = threadIdx.x;
  const int lane = tid & 31;
  const int wave = tid >> 5;
  const int m0 = blockIdx.y * BM;
  const int n0 = blockIdx.x * BN;
  const int mt  = wave >> 1;        // 0..3
  const int nt0 = (wave & 1) * 2;   // 0 or 2 (two adjacent 16-wide tiles)

  auto stage = [&](int buf, int k0) {
    if constexpr (A_BF16) {
#if HAVE_ASYNC_COPY
      // One 16-byte async copy per thread: 2048 bf16 = 256 thr * 8 elems.
      {
        const int r = tid >> 2, c = (tid & 3) * 8;
        async_copy16(&As[buf][r][c], &A[(size_t)(m0 + r) * K + (k0 + c)]);
      }
#else
      for (int i = tid; i < BM * BK; i += 256) {
        int r = i >> 5, c = i & 31;
        As[buf][r][c] = A[(size_t)(m0 + r) * K + (k0 + c)];
      }
#endif
    } else {
      for (int i = tid; i < BM * BK; i += 256) {
        int r = i >> 5, c = i & 31;
        As[buf][r][c] = (bf16_t)(float)A[(size_t)(m0 + r) * K + (k0 + c)];
      }
    }
    for (int i = tid; i < BK * BN; i += 256) {
      int r = i >> 6, c = i & 63;
      Bs[buf][r][c] = (bf16_t)B[(size_t)(k0 + r) * N + (n0 + c)];
    }
    // Prefetch the chunk after next (gfx1250 global_prefetch_b8).
    if (k0 + 2 * BK <= K - BK) {
      __builtin_prefetch(&A[(size_t)(m0 + (tid >> 2)) * K + (k0 + 2 * BK)], 0, 0);
      if (tid < BK)
        __builtin_prefetch(&B[(size_t)(k0 + 2 * BK + tid) * N + n0], 0, 0);
    }
  };

  stage(0, 0);
  if (A_BF16) async_wait0();
  __syncthreads();

  v8f acc0 = {0.f, 0.f, 0.f, 0.f, 0.f, 0.f, 0.f, 0.f};
  v8f acc1 = {0.f, 0.f, 0.f, 0.f, 0.f, 0.f, 0.f, 0.f};
  const int nk = K / BK;
  for (int kc = 0; kc < nk; ++kc) {
    const int buf = kc & 1;
    if (kc + 1 < nk) stage(buf ^ 1, (kc + 1) * BK);
    v16bf a  = load_a_frag(&As[buf][mt * 16][0], BK, lane);
    v16bf b0 = load_b_frag(&Bs[buf][0][nt0 * 16], BN, lane);
    v16bf b1 = load_b_frag(&Bs[buf][0][(nt0 + 1) * 16], BN, lane);
    acc0 = wmma_bf16(a, b0, acc0);
    acc1 = wmma_bf16(a, b1, acc1);
    if (A_BF16) async_wait0();
    __syncthreads();
  }

  const int n  = lane & 15;
  const int mb = (lane >> 4) * 8;
  const int gm = m0 + mt * 16 + mb;
#pragma unroll
  for (int t = 0; t < 2; ++t) {
    const int gn = n0 + (nt0 + t) * 16 + n;
#pragma unroll
    for (int j = 0; j < 8; ++j) {
      float v = (t == 0) ? acc0[j] : acc1[j];
      if (SIGMOID) v = 1.0f / (1.0f + __expf(-v));
      Y[(size_t)(gm + j) * N + gn] = (OT)v;
    }
  }
}

// ---------------------------------------------------------------------------
// Pair bias: bias[h, i, j] = sum_z pair[(i*256+j), z] * Wb[z, h].   (N=8 < 16,
// skinny -> VALU; Wb staged in LDS.)  Grid: 256 blocks x 256 thr, 1 thr per ij.
// ---------------------------------------------------------------------------
__global__ __launch_bounds__(256) void pair_bias_kernel(
    const float* __restrict__ pair, const float* __restrict__ Wb,
    float* __restrict__ biasb) {
  __shared__ float bp[128 * 8];
  for (int i = threadIdx.x; i < 128 * 8; i += 256) bp[i] = Wb[i];
  __syncthreads();

  const int ij = blockIdx.x * 256 + threadIdx.x;  // 0..65535
  const float* p = pair + (size_t)ij * 128;
  float acc[8] = {0.f, 0.f, 0.f, 0.f, 0.f, 0.f, 0.f, 0.f};
  for (int z = 0; z < 128; ++z) {
    const float pv = p[z];
#pragma unroll
    for (int h = 0; h < 8; ++h) acc[h] = fmaf(pv, bp[z * 8 + h], acc[h]);
  }
#pragma unroll
  for (int h = 0; h < 8; ++h) biasb[(size_t)h * 65536 + ij] = acc[h];
}

// ---------------------------------------------------------------------------
// Attention: one block per (s, h, 64-query tile).  256 thr = 8 waves.
//   S = (Q*scale) @ K^T          (64x256, one WMMA per 16x16 tile since C=32)
//   P = softmax_row(S) + bias    (bias added AFTER softmax, per reference)
//   O = (P @ V) * gate -> bf16 attend buffer
// LDS: 4 + 16 + 16 + 64 + 32 = 132 KB.
// ---------------------------------------------------------------------------
__global__ __launch_bounds__(256) void attn_kernel(
    const bf16_t* __restrict__ QKV,    // [32768, 768] bf16 (q|k|v, head-major c)
    const float* __restrict__ gate,    // [32768, 256] f32
    const float* __restrict__ biasb,   // [8, 256, 256] f32
    bf16_t* __restrict__ attnb) {      // [32768, 256] bf16
  constexpr int R = 256, HD = 32, QT = 64;
  const int qt = blockIdx.x, h = blockIdx.y, s = blockIdx.z;
  const int tid = threadIdx.x, lane = tid & 31, wave = tid >> 5;

  __shared__ __align__(16) bf16_t Qs[QT][HD];
  __shared__ __align__(16) bf16_t Ks[R][HD];
  __shared__ __align__(16) bf16_t Vs[R][HD];
  __shared__ __align__(16) float  Ss[QT][R];
  __shared__ __align__(16) bf16_t Ps[QT][R];

  // K/V: one 32-element row per thread (64B), conversion-free staging.
  {
    const bf16_t* kp = QKV + ((size_t)(s * R + tid) * 768 + 256 + h * HD);
#if HAVE_ASYNC_COPY
#pragma unroll
    for (int i = 0; i < 4; ++i) {
      async_copy16(&Ks[tid][i * 8], kp + i * 8);
      async_copy16(&Vs[tid][i * 8], kp + 256 + i * 8);
    }
#else
    const uint4* ksrc = (const uint4*)kp;
    const uint4* vsrc = (const uint4*)(kp + 256);
    uint4* kd = (uint4*)&Ks[tid][0];
    uint4* vd = (uint4*)&Vs[tid][0];
#pragma unroll
    for (int i = 0; i < 4; ++i) { kd[i] = ksrc[i]; vd[i] = vsrc[i]; }
#endif
  }
  // Q tile, with softmax scale folded in (needs conversion -> VALU path).
  for (int i = tid; i < QT * HD; i += 256) {
    const int r = i >> 5, c = i & 31;
    const float q =
        (float)QKV[(size_t)(s * R + qt * QT + r) * 768 + h * HD + c];
    Qs[r][c] = (bf16_t)(q * 0.17677669529663687f);  // 1/sqrt(32)
  }
  async_wait0();
  __syncthreads();

  // S = Q @ K^T : 4x16 tiles of 16x16; 8 tiles per wave, each a single WMMA.
  {
    const int n = lane & 15, mb = (lane >> 4) * 8;
    for (int t = wave; t < 64; t += 8) {
      const int mt = t >> 4, ntt = t & 15;
      v16bf a = load_a_frag(&Qs[mt * 16][0], HD, lane);
      v16bf b = load_bT_frag(&Ks[ntt * 16][0], HD, lane);
      v8f c = {0.f, 0.f, 0.f, 0.f, 0.f, 0.f, 0.f, 0.f};
      c = wmma_bf16(a, b, c);
#pragma unroll
      for (int j = 0; j < 8; ++j)
        Ss[mt * 16 + mb + j][ntt * 16 + n] = c[j];
    }
  }
  __syncthreads();

  // Row softmax (wave32 butterfly reductions), then += bias, store bf16 P.
  {
    const int r0 = wave * 8;
    for (int rr = 0; rr < 8; ++rr) {
      const int r = r0 + rr;
      float vals[8];
      float mx = -3.0e38f;
#pragma unroll
      for (int i = 0; i < 8; ++i) {
        vals[i] = Ss[r][lane + 32 * i];
        mx = fmaxf(mx, vals[i]);
      }
#pragma unroll
      for (int off = 16; off > 0; off >>= 1)
        mx = fmaxf(mx, __shfl_xor(mx, off, 32));
      float sum = 0.f;
#pragma unroll
      for (int i = 0; i < 8; ++i) {
        vals[i] = __expf(vals[i] - mx);
        sum += vals[i];
      }
#pragma unroll
      for (int off = 16; off > 0; off >>= 1) sum += __shfl_xor(sum, off, 32);
      const float inv = 1.0f / sum;
      const int gq = qt * QT + r;
      const float* brow = biasb + ((size_t)h * 256 + gq) * 256;
#pragma unroll
      for (int i = 0; i < 8; ++i) {
        const int cidx = lane + 32 * i;
        Ps[r][cidx] = (bf16_t)(vals[i] * inv + brow[cidx]);
      }
    }
  }
  __syncthreads();

  // O = P @ V : 4x2 tiles of 16x16, one per wave; fully unrolled 8 WMMAs.
  {
    const int mt = wave >> 1, ntt = wave & 1;
    v8f acc = {0.f, 0.f, 0.f, 0.f, 0.f, 0.f, 0.f, 0.f};
#pragma unroll
    for (int kc = 0; kc < R; kc += 32) {
      v16bf a = load_a_frag(&Ps[mt * 16][kc], R, lane);
      v16bf b = load_b_frag(&Vs[kc][ntt * 16], HD, lane);
      acc = wmma_bf16(a, b, acc);
    }
    const int n = lane & 15, mb = (lane >> 4) * 8;
    const int gr = s * R + qt * QT + mt * 16 + mb;
    const int gc = h * HD + ntt * 16 + n;
#pragma unroll
    for (int j = 0; j < 8; ++j) {
      const size_t idx = (size_t)(gr + j) * 256 + gc;
      attnb[idx] = (bf16_t)(acc[j] * gate[idx]);
    }
  }
}

// ---------------------------------------------------------------------------
// Host launcher.
// Inputs: msa[1,128,256,256] pair[1,256,256,128] Wg[256,256] Wqkv[256,768]
//         Wout[256,256] Wb[128,8]  (all f32).  Output: [1,128,256,256] f32.
// Workspace: qkv bf16 48MB | gate f32 32MB | bias f32 2MB | attend bf16 16MB.
// ---------------------------------------------------------------------------
extern "C" void kernel_launch(void* const* d_in, const int* in_sizes, int n_in,
                              void* d_out, int out_size, void* d_ws,
                              size_t ws_size, hipStream_t stream) {
  const float* msa  = (const float*)d_in[0];
  const float* pair = (const float*)d_in[1];
  const float* Wg   = (const float*)d_in[2];
  const float* Wqkv = (const float*)d_in[3];
  const float* Wout = (const float*)d_in[4];
  const float* Wb   = (const float*)d_in[5];
  float* out = (float*)d_out;

  const int ROWS = 32768;  // S*R
  char* ws = (char*)d_ws;
  const size_t QKV_BYTES  = (size_t)ROWS * 768 * sizeof(bf16_t);  // 48 MB
  const size_t GATE_BYTES = (size_t)ROWS * 256 * sizeof(float);   // 32 MB
  const size_t BIAS_BYTES = (size_t)8 * 65536 * sizeof(float);    //  2 MB
  bf16_t* qkv   = (bf16_t*)ws;
  float*  gate  = (float*)(ws + QKV_BYTES);
  float*  biasb = (float*)(ws + QKV_BYTES + GATE_BYTES);
  bf16_t* attnb = (bf16_t*)(ws + QKV_BYTES + GATE_BYTES + BIAS_BYTES);

  // 1) gate = sigmoid(X @ Wg)            [32768,256]
  gemm_wmma_kernel<float, float, true>
      <<<dim3(256 / 64, ROWS / 64), 256, 0, stream>>>(msa, Wg, gate, ROWS, 256, 256);
  // 2) qkv = X @ Wqkv  (bf16 out)        [32768,768]
  gemm_wmma_kernel<float, bf16_t, false>
      <<<dim3(768 / 64, ROWS / 64), 256, 0, stream>>>(msa, Wqkv, qkv, ROWS, 768, 256);
  // 3) pair bias                         [8,256,256]
  pair_bias_kernel<<<256, 256, 0, stream>>>(pair, Wb, biasb);
  // 4) attention per (s, h, q-tile)
  attn_kernel<<<dim3(4, 8, 128), 256, 0, stream>>>(qkv, gate, biasb, attnb);
  // 5) out = attend @ Wout               [32768,256] f32
  gemm_wmma_kernel<bf16_t, float, false>
      <<<dim3(256 / 64, ROWS / 64), 256, 0, stream>>>(attnb, Wout, out, ROWS, 256, 256);
}
